// NormLinearAttention_45165876085087
// MI455X (gfx1250) — matrix operand order repaired and checked by reading, not compiled, not run on hardware
//
#include <hip/hip_runtime.h>
#include <cstdint>

// ---------------------------------------------------------------------------
// NormLinearAttention for MI455X (gfx1250, wave32, WMMA + async-to-LDS).
// B=4, N=4096, DIM=1024, HEADS=16, DHEAD=64, INNER=1024.
// ---------------------------------------------------------------------------

typedef _Float16 half_t;
typedef __attribute__((ext_vector_type(16))) _Float16 v16h;
typedef __attribute__((ext_vector_type(8)))  _Float16 v8h;
typedef __attribute__((ext_vector_type(8)))  float    v8f;

#define WMMA_F16(a, b, c) \
  __builtin_amdgcn_wmma_f32_16x16x32_f16(false, (a), false, (b), (short)0, (c), false, false)

// Async DMA: copy 16 bytes global -> LDS, tracked by ASYNCcnt.
// lds_off = low 32 bits of the generic pointer (wave-relative LDS address per
// the gfx1250 flat-aperture mapping).
static __device__ __forceinline__ void async_copy_b128(unsigned lds_off, const void* gptr) {
  asm volatile("global_load_async_to_lds_b128 %0, %1, off"
               :
               : "v"(lds_off), "v"((unsigned long long)(uintptr_t)gptr)
               : "memory");
}
static __device__ __forceinline__ void wait_async() {
  asm volatile("s_wait_asynccnt 0x0" ::: "memory");
}
static __device__ __forceinline__ unsigned lds_addr(const void* p) {
  return (unsigned)(uintptr_t)p;
}

// A fragment (16x32 f16): lane = half-group h (lane>>4) + row m (lane&15).
// j=0..7 -> K = k0+8h+j ; j=8..15 -> K = k0+16+8h+(j-8). 16B-aligned ds_load_b128s.
static __device__ __forceinline__ v16h frag_a(const half_t* s, int mbase, int k0,
                                              int lda, int lane) {
  const half_t* p = s + (mbase + (lane & 15)) * lda + k0 + 8 * (lane >> 4);
  v8h lo = *(const v8h*)(p);
  v8h hi = *(const v8h*)(p + 16);
  v16h r;
#pragma unroll
  for (int j = 0; j < 8; ++j) { r[j] = lo[j]; r[8 + j] = hi[j]; }
  return r;
}

// B fragment (32x16 f16), tile staged in LDS as [n][k]:
// lane n = lane&15, j=0..7 -> K = k0+8h+j ; j=8..15 -> K = k0+16+8h+(j-8)
static __device__ __forceinline__ v16h frag_b(const half_t* s, int nbase, int k0,
                                              int lda, int lane) {
  const half_t* p = s + (nbase + (lane & 15)) * lda + k0 + 8 * (lane >> 4);
  v8h lo = *(const v8h*)(p);
  v8h hi = *(const v8h*)(p + 16);
  v16h r;
#pragma unroll
  for (int j = 0; j < 8; ++j) { r[j] = lo[j]; r[8 + j] = hi[j]; }
  return r;
}

// ---------------------------------------------------------------------------
// Prep 0a: elementwise f32 -> f16 (x -> X16). 8 elements/thread.
// ---------------------------------------------------------------------------
__global__ __launch_bounds__(256) void cvt_f32_f16(const float* __restrict__ in,
                                                   half_t* __restrict__ out) {
  size_t i = ((size_t)blockIdx.x * 256 + threadIdx.x) * 8;
  float4 a = *(const float4*)(in + i);
  float4 b = *(const float4*)(in + i + 4);
  v8h o;
  o[0] = (half_t)a.x; o[1] = (half_t)a.y; o[2] = (half_t)a.z; o[3] = (half_t)a.w;
  o[4] = (half_t)b.x; o[5] = (half_t)b.y; o[6] = (half_t)b.z; o[7] = (half_t)b.w;
  *(v8h*)(out + i) = o;
}

// ---------------------------------------------------------------------------
// Prep 0b: transpose f32 [R][C] -> f16 [C][R] (weights into B-ready layout).
// 32x32 tiles through LDS; block = 256 threads (32x8).
// ---------------------------------------------------------------------------
__global__ __launch_bounds__(256) void transpose_f32_f16(
    const float* __restrict__ in, half_t* __restrict__ out, int R, int C) {
  __shared__ float t[32][33];
  const int c0 = blockIdx.x * 32, r0 = blockIdx.y * 32;
  const int tx = threadIdx.x & 31, ty = threadIdx.x >> 5;   // 32 x 8
#pragma unroll
  for (int j = 0; j < 32; j += 8)
    t[ty + j][tx] = in[(size_t)(r0 + ty + j) * C + c0 + tx];
  __syncthreads();
#pragma unroll
  for (int j = 0; j < 32; j += 8)
    out[(size_t)(c0 + ty + j) * R + r0 + tx] = (half_t)t[tx][ty + j];
}

// ---------------------------------------------------------------------------
// Kernel 1: qkv = x @ w_qkv  (16384x1024 @ 1024x3072), f16 WMMA, async staging.
// Outputs: Q as [b,h,n,d]; K,V pre-transposed as [b,h,d,n] (ready for kernel 2).
// Tile 128x128x32, 8 waves; wave computes 64x32 = 4x2 WMMA tiles.
// ---------------------------------------------------------------------------
__global__ __launch_bounds__(256) void qkv_gemm_wmma(
    const half_t* __restrict__ X16, const half_t* __restrict__ Wt,  // Wt:[3072][1024]
    half_t* __restrict__ Qb, half_t* __restrict__ Kt, half_t* __restrict__ Vt) {
  constexpr int KD = 1024, LDA = 40, LDB = 40;
  __shared__ __align__(16) half_t sA[128 * LDA];
  __shared__ __align__(16) half_t sB[128 * LDB];
  const int tid  = threadIdx.x;
  const int lane = tid & 31;
  const int wave = tid >> 5;
  const int row0 = blockIdx.x * 128;
  const int col0 = blockIdx.y * 128;
  const int wm = (wave & 1) * 64;
  const int wn = (wave >> 1) * 32;
  const unsigned sA0 = lds_addr(&sA[0]);
  const unsigned sB0 = lds_addr(&sB[0]);

  v8f acc[4][2];
#pragma unroll
  for (int i = 0; i < 4; ++i)
#pragma unroll
    for (int j = 0; j < 2; ++j)
#pragma unroll
      for (int r = 0; r < 8; ++r) acc[i][j][r] = 0.0f;

  for (int k0 = 0; k0 < KD; k0 += 32) {
    if (k0 + 32 < KD)
      __builtin_prefetch(X16 + (size_t)(row0 + (tid >> 1)) * KD + k0 + 32, 0, 0);
    // async stage: A tile 128 rows x 32 halves, B tile 128 n-rows x 32 halves.
    // 512 16B chunks each -> 2 async copies per matrix per thread.
#pragma unroll
    for (int u = 0; u < 2; ++u) {
      int idx = tid + u * 256;
      int r = idx >> 2, c = idx & 3;
      async_copy_b128(sA0 + (unsigned)(r * LDA + c * 8) * 2,
                      X16 + (size_t)(row0 + r) * KD + k0 + c * 8);
      async_copy_b128(sB0 + (unsigned)(r * LDB + c * 8) * 2,
                      Wt + (size_t)(col0 + r) * KD + k0 + c * 8);
    }
    wait_async();
    __syncthreads();
    v16h af[4], bf[2];
#pragma unroll
    for (int mt = 0; mt < 4; ++mt) af[mt] = frag_a(sA, wm + mt * 16, 0, LDA, lane);
#pragma unroll
    for (int nt = 0; nt < 2; ++nt) bf[nt] = frag_b(sB, wn + nt * 16, 0, LDB, lane);
#pragma unroll
    for (int mt = 0; mt < 4; ++mt)
#pragma unroll
      for (int nt = 0; nt < 2; ++nt)
        acc[mt][nt] = WMMA_F16(af[mt], bf[nt], acc[mt][nt]);
    __syncthreads();
  }

  // Epilogue: part is UNIFORM per block (128-wide tile never crosses a 1024
  // boundary), so these branches are scalar.
  const int part = col0 >> 10;                 // 0=q 1=k 2=v
  const int hi = lane >> 4, ln16 = lane & 15;
#pragma unroll
  for (int mt = 0; mt < 4; ++mt)
#pragma unroll
    for (int nt = 0; nt < 2; ++nt)
#pragma unroll
      for (int r = 0; r < 8; ++r) {
        int m = row0 + wm + mt * 16 + r + 8 * hi;   // global row (b*4096+n)
        int c = col0 + wn + nt * 16 + ln16;
        float v = acc[mt][nt][r];
        int inner = c & 1023;
        int hh = inner >> 6, dd = inner & 63;
        int bb = m >> 12, nn = m & 4095;
        int bh = bb * 16 + hh;
        if (part == 0) {
          Qb[((size_t)bh * 4096 + nn) * 64 + dd] = (half_t)fmaxf(v, 0.0f);
        } else if (part == 1) {
          Kt[((size_t)bh * 64 + dd) * 4096 + nn] = (half_t)fmaxf(v, 0.0f);
        } else {
          Vt[((size_t)bh * 64 + dd) * 4096 + nn] = (half_t)v;
        }
      }
}

// ---------------------------------------------------------------------------
// Kernel 2: kv[b,h] = K^T @ V  (64x4096 @ 4096x64 -> 64x64).
// K,V arrive pre-transposed [d][n], so tiles are straight async copies.
// One block per (b,h); 16 output tiles, 2 per wave. Output stored [dv][dk].
// ---------------------------------------------------------------------------
__global__ __launch_bounds__(256) void kv_gemm_wmma(
    const half_t* __restrict__ Kt, const half_t* __restrict__ Vt,
    half_t* __restrict__ KVt) {
  constexpr int LDA = 40;
  __shared__ __align__(16) half_t sA[64 * LDA];   // K^T slice [dk][n]
  __shared__ __align__(16) half_t sB[64 * LDA];   // V^T slice [dv][n]
  const int tid = threadIdx.x, lane = tid & 31, wave = tid >> 5;
  const int bh = blockIdx.x;
  const half_t* Kh = Kt + (size_t)bh * 64 * 4096;
  const half_t* Vh = Vt + (size_t)bh * 64 * 4096;
  const int mt  = wave >> 1;
  const int nt0 = (wave & 1) * 2;
  const unsigned sA0 = lds_addr(&sA[0]);
  const unsigned sB0 = lds_addr(&sB[0]);
  const int dk = tid >> 2, cc = tid & 3;          // 256 16B chunks per matrix

  v8f acc[2];
#pragma unroll
  for (int t = 0; t < 2; ++t)
#pragma unroll
    for (int r = 0; r < 8; ++r) acc[t][r] = 0.0f;

  for (int n0 = 0; n0 < 4096; n0 += 32) {
    async_copy_b128(sA0 + (unsigned)(dk * LDA + cc * 8) * 2,
                    Kh + (size_t)dk * 4096 + n0 + cc * 8);
    async_copy_b128(sB0 + (unsigned)(dk * LDA + cc * 8) * 2,
                    Vh + (size_t)dk * 4096 + n0 + cc * 8);
    wait_async();
    __syncthreads();
    v16h a  = frag_a(sA, mt * 16, 0, LDA, lane);
    v16h b0 = frag_b(sB, nt0 * 16, 0, LDA, lane);
    v16h b1 = frag_b(sB, nt0 * 16 + 16, 0, LDA, lane);
    acc[0] = WMMA_F16(a, b0, acc[0]);
    acc[1] = WMMA_F16(a, b1, acc[1]);
    __syncthreads();
  }

  const int hi = lane >> 4, ln16 = lane & 15;
#pragma unroll
  for (int t = 0; t < 2; ++t)
#pragma unroll
    for (int r = 0; r < 8; ++r) {
      int m = mt * 16 + r + 8 * hi;            // dk
      int c = (nt0 + t) * 16 + ln16;           // dv
      KVt[(size_t)bh * 4096 + c * 64 + m] = (half_t)acc[t][r];   // [dv][dk]
    }
}

// ---------------------------------------------------------------------------
// Kernel 3: o[b,h] = Q[b,h] @ kv[b,h]  (4096x64 @ 64x64), f32 into
// [b, n, h*64+dv] layout for LayerNorm. kv arrives [dv][dk] -> B-ready.
// ---------------------------------------------------------------------------
__global__ __launch_bounds__(256) void o_gemm_wmma(
    const half_t* __restrict__ Qb, const half_t* __restrict__ KVt,
    float* __restrict__ Ob) {
  constexpr int LDA = 72;
  __shared__ __align__(16) half_t sA[128 * LDA];  // Q tile [row][dk]
  __shared__ __align__(16) half_t sB[64 * LDA];   // kv [dv][dk]
  const int tid = threadIdx.x, lane = tid & 31, wave = tid >> 5;
  const int bh = blockIdx.y;
  const int row0 = blockIdx.x * 128;
  const half_t* Qh  = Qb  + (size_t)bh * 4096 * 64;
  const half_t* kvp = KVt + (size_t)bh * 4096;
  const unsigned sA0 = lds_addr(&sA[0]);
  const unsigned sB0 = lds_addr(&sB[0]);

#pragma unroll
  for (int u = 0; u < 4; ++u) {                 // 1024 chunks: 128 rows x 8
    int idx = tid + u * 256;
    int r = idx >> 3, cq = idx & 7;
    async_copy_b128(sA0 + (unsigned)(r * LDA + cq * 8) * 2,
                    Qh + (size_t)(row0 + r) * 64 + cq * 8);
  }
#pragma unroll
  for (int u = 0; u < 2; ++u) {                 // 512 chunks: 64 rows x 8
    int idx = tid + u * 256;
    int dv = idx >> 3, cq = idx & 7;
    async_copy_b128(sB0 + (unsigned)(dv * LDA + cq * 8) * 2,
                    kvp + (size_t)dv * 64 + cq * 8);
  }
  wait_async();
  __syncthreads();

  v8f acc[4];
#pragma unroll
  for (int t = 0; t < 4; ++t)
#pragma unroll
    for (int r = 0; r < 8; ++r) acc[t][r] = 0.0f;

#pragma unroll
  for (int k0 = 0; k0 < 64; k0 += 32) {
    v16h a = frag_a(sA, wave * 16, k0, LDA, lane);
#pragma unroll
    for (int nt = 0; nt < 4; ++nt) {
      v16h b = frag_b(sB, nt * 16, k0, LDA, lane);
      acc[nt] = WMMA_F16(a, b, acc[nt]);
    }
  }

  const int hi = lane >> 4, ln16 = lane & 15;
  const int b = bh >> 4, h = bh & 15;
#pragma unroll
  for (int nt = 0; nt < 4; ++nt)
#pragma unroll
    for (int r = 0; r < 8; ++r) {
      int m  = row0 + wave * 16 + r + 8 * hi;
      int dv = nt * 16 + ln16;
      Ob[(size_t)(b * 4096 + m) * 1024 + h * 64 + dv] = acc[nt][r];
    }
}

// ---------------------------------------------------------------------------
// Kernel 4: LayerNorm over inner=1024 per row, f32 in, f16 out.
// ---------------------------------------------------------------------------
__global__ __launch_bounds__(256) void ln_wmma_prep(
    const float* __restrict__ Ob, const float* __restrict__ gamma,
    const float* __restrict__ beta, half_t* __restrict__ XNb) {
  __shared__ float red[256];
  const int tid = threadIdx.x;
  const size_t row = blockIdx.x;
  float4 v = *(const float4*)(Ob + row * 1024 + tid * 4);
  red[tid] = v.x + v.y + v.z + v.w;
  __syncthreads();
  for (int off = 128; off > 0; off >>= 1) {
    if (tid < off) red[tid] += red[tid + off];
    __syncthreads();
  }
  float mean = red[0] * (1.0f / 1024.0f);
  __syncthreads();
  float dx = v.x - mean, dy = v.y - mean, dz = v.z - mean, dw = v.w - mean;
  red[tid] = dx * dx + dy * dy + dz * dz + dw * dw;
  __syncthreads();
  for (int off = 128; off > 0; off >>= 1) {
    if (tid < off) red[tid] += red[tid + off];
    __syncthreads();
  }
  float rinv = rsqrtf(red[0] * (1.0f / 1024.0f) + 1e-5f);
  int c = tid * 4;
  float4 g  = *(const float4*)(gamma + c);
  float4 be = *(const float4*)(beta + c);
  half_t* o = XNb + row * 1024 + c;
  o[0] = (half_t)(dx * rinv * g.x + be.x);
  o[1] = (half_t)(dy * rinv * g.y + be.y);
  o[2] = (half_t)(dz * rinv * g.z + be.z);
  o[3] = (half_t)(dw * rinv * g.w + be.w);
}

// ---------------------------------------------------------------------------
// Kernel 5: out = xn @ w_out + b_out  (16384x1024 @ 1024x1024), f32 result.
// Both operands f16 row-major in B-ready layouts -> pure async staging.
// ---------------------------------------------------------------------------
__global__ __launch_bounds__(256) void out_gemm_wmma(
    const half_t* __restrict__ XNb, const half_t* __restrict__ Wt,  // Wt:[1024][1024]
    const float* __restrict__ bout, float* __restrict__ Out) {
  constexpr int KD = 1024, ND = 1024, LDA = 40, LDB = 40;
  __shared__ __align__(16) half_t sA[128 * LDA];
  __shared__ __align__(16) half_t sB[128 * LDB];
  const int tid = threadIdx.x, lane = tid & 31, wave = tid >> 5;
  const int row0 = blockIdx.x * 128;
  const int col0 = blockIdx.y * 128;
  const int wm = (wave & 1) * 64;
  const int wn = (wave >> 1) * 32;
  const unsigned sA0 = lds_addr(&sA[0]);
  const unsigned sB0 = lds_addr(&sB[0]);

  v8f acc[4][2];
#pragma unroll
  for (int i = 0; i < 4; ++i)
#pragma unroll
    for (int j = 0; j < 2; ++j)
#pragma unroll
      for (int r = 0; r < 8; ++r) acc[i][j][r] = 0.0f;

  for (int k0 = 0; k0 < KD; k0 += 32) {
    if (k0 + 32 < KD)
      __builtin_prefetch(XNb + (size_t)(row0 + (tid >> 1)) * KD + k0 + 32, 0, 0);
#pragma unroll
    for (int u = 0; u < 2; ++u) {
      int idx = tid + u * 256;
      int r = idx >> 2, c = idx & 3;
      async_copy_b128(sA0 + (unsigned)(r * LDA + c * 8) * 2,
                      XNb + (size_t)(row0 + r) * KD + k0 + c * 8);
      async_copy_b128(sB0 + (unsigned)(r * LDB + c * 8) * 2,
                      Wt + (size_t)(col0 + r) * KD + k0 + c * 8);
    }
    wait_async();
    __syncthreads();
    v16h af[4], bf[2];
#pragma unroll
    for (int mt = 0; mt < 4; ++mt) af[mt] = frag_a(sA, wm + mt * 16, 0, LDA, lane);
#pragma unroll
    for (int nt = 0; nt < 2; ++nt) bf[nt] = frag_b(sB, wn + nt * 16, 0, LDB, lane);
#pragma unroll
    for (int mt = 0; mt < 4; ++mt)
#pragma unroll
      for (int nt = 0; nt < 2; ++nt)
        acc[mt][nt] = WMMA_F16(af[mt], bf[nt], acc[mt][nt]);
    __syncthreads();
  }

  const int hi = lane >> 4, ln16 = lane & 15;
#pragma unroll
  for (int mt = 0; mt < 4; ++mt)
#pragma unroll
    for (int nt = 0; nt < 2; ++nt)
#pragma unroll
      for (int r = 0; r < 8; ++r) {
        int m = row0 + wm + mt * 16 + r + 8 * hi;
        int c = col0 + wn + nt * 16 + ln16;
        Out[(size_t)m * ND + c] = acc[mt][nt][r] + bout[c];
      }
}

// ---------------------------------------------------------------------------
// Host launcher
// ---------------------------------------------------------------------------
extern "C" void kernel_launch(void* const* d_in, const int* in_sizes, int n_in,
                              void* d_out, int out_size, void* d_ws, size_t ws_size,
                              hipStream_t stream) {
  const float* x     = (const float*)d_in[0];  // [4,4096,1024]
  const float* w_qkv = (const float*)d_in[1];  // [1024,3072]
  const float* gamma = (const float*)d_in[2];  // [1024]
  const float* beta  = (const float*)d_in[3];  // [1024]
  const float* w_out = (const float*)d_in[4];  // [1024,1024]
  const float* b_out = (const float*)d_in[5];  // [1024]
  float* out = (float*)d_out;                  // [4,4096,1024]

  const size_t NTOK = (size_t)16384 * 1024;    // 16.7M elements
  char* ws = (char*)d_ws;
  half_t* X16   = (half_t*)ws;                          // f16 [16384][1024]
  half_t* Wq16t = X16 + NTOK;                           // f16 [3072][1024]
  half_t* Wo16t = Wq16t + (size_t)3072 * 1024;          // f16 [1024][1024]
  half_t* Qb    = Wo16t + (size_t)1024 * 1024;          // f16 [b,h,n,d]
  half_t* Kt    = Qb + NTOK;                            // f16 [b,h,d,n]
  half_t* Vt    = Kt + NTOK;                            // f16 [b,h,d,n]
  half_t* KVt   = Vt + NTOK;                            // f16 [bh][dv][dk]
  float*  Ob    = (float*)(KVt + (size_t)64 * 64 * 64); // f32 [16384][1024]
  half_t* XNb   = (half_t*)(Ob + NTOK);                 // f16 [16384][1024]

  cvt_f32_f16<<<dim3(8192), 256, 0, stream>>>(x, X16);
  transpose_f32_f16<<<dim3(96, 32), 256, 0, stream>>>(w_qkv, Wq16t, 1024, 3072);
  transpose_f32_f16<<<dim3(32, 32), 256, 0, stream>>>(w_out, Wo16t, 1024, 1024);
  qkv_gemm_wmma<<<dim3(128, 24), 256, 0, stream>>>(X16, Wq16t, Qb, Kt, Vt);
  kv_gemm_wmma<<<dim3(64), 256, 0, stream>>>(Kt, Vt, KVt);
  o_gemm_wmma<<<dim3(32, 64), 256, 0, stream>>>(Qb, KVt, Ob);
  ln_wmma_prep<<<dim3(16384), 256, 0, stream>>>(Ob, gamma, beta, XNb);
  out_gemm_wmma<<<dim3(128, 8), 256, 0, stream>>>(XNb, Wo16t, b_out, out);
}
